// CausalSelfAttention_11639361372783
// MI455X (gfx1250) — compile-verified, matrix operand order checked by reading, not run on hardware
//
#include <hip/hip_runtime.h>
#include <hip/hip_bf16.h>

// ---------------- problem constants ----------------
#define N_HEAD 16
#define N_EMBD 1024
#define SEQ    2048
#define BATCH  4
#define HDIM   64            // N_EMBD / N_HEAD
#define MROWS  (BATCH * SEQ) // 8192
#define KDIM   1024          // inner dim of both GEMMs

typedef __attribute__((ext_vector_type(16))) __bf16       v16bf;
typedef __attribute__((ext_vector_type(8)))  float        v8f;
typedef __attribute__((ext_vector_type(4)))  float        f4;
typedef __attribute__((ext_vector_type(4)))  unsigned int u4;

union frag_t { v16bf v; u4 q[2]; };

__device__ __forceinline__ unsigned pk2(float a, float b) {
    __bf16 lo = (__bf16)a, hi = (__bf16)b;          // v_cvt_pk_bf16_f32
    unsigned short ls = __builtin_bit_cast(unsigned short, lo);
    unsigned short hs = __builtin_bit_cast(unsigned short, hi);
    return (unsigned)ls | ((unsigned)hs << 16);
}
__device__ __forceinline__ u4 pk8(f4 a, f4 b) {
    u4 r;
    r.x = pk2(a.x, a.y); r.y = pk2(a.z, a.w);
    r.z = pk2(b.x, b.y); r.w = pk2(b.z, b.w);
    return r;
}

// =====================================================================
// Tiled bf16 WMMA GEMM:  out[M,N] = A[M,K] @ B[K,N] + bias[N]
// MODE 0: scatter columns into q/k/v workspace laid out [B,H,T,D]
// MODE 1: plain row-major store into outp
// 256 threads = 8 waves (4x2); macro tile 128(M) x 128(N), K-step 32,
// double-buffered LDS (one barrier per K-step).
// Each wave owns 32x64 = 2x4 WMMA accumulators (8 wmma / K-step).
// =====================================================================
#define TM 128
#define TN 128
#define TK 32
#define LDP 40   // LDS pitch (elements): 80B rows keep 16B-aligned chunks

template<int MODE, int NDIM>
__global__ __launch_bounds__(256)
void gemm_bf16_wmma(const float* __restrict__ A, const float* __restrict__ Bw,
                    const float* __restrict__ bias,
                    float* __restrict__ qw, float* __restrict__ kw,
                    float* __restrict__ vw, float* __restrict__ outp)
{
    __shared__ __bf16 As[2][TM][LDP];
    __shared__ __bf16 Bs[2][TN][LDP];   // [n][k]  (B transposed in LDS)

    const int tid  = threadIdx.x;
    const int lane = tid & 31;
    const int wave = tid >> 5;
    const int wm   = (wave & 3) * 32;   // wave row offset in tile
    const int wn   = (wave >> 2) * 64;  // wave col offset in tile
    const int r16  = lane & 15;
    const int klo  = (lane < 16) ? 0 : 8;
    const int hb   = (lane < 16) ? 0 : 8;

    const int n0 = blockIdx.x * TN;
    const int m0 = blockIdx.y * TM;

    const int bn  = tid & 127;          // B staging: fixed column
    const int bk0 = (tid >> 7) * 16;

    auto stage = [&](int buf, int k0) {
        // A: 2 chunks of 8 consecutive floats per thread
        #pragma unroll
        for (int j = 0; j < 2; ++j) {
            int c   = tid + j * 256;
            int row = c >> 2;
            int col = (c & 3) * 8;
            const f4* gp = (const f4*)(A + (size_t)(m0 + row) * KDIM + k0 + col);
            *(u4*)&As[buf][row][col] = pk8(gp[0], gp[1]);
        }
        // B transposed: walk K for fixed n (lane-coalesced in n)
        float t[16];
        #pragma unroll
        for (int j = 0; j < 16; ++j)
            t[j] = Bw[(size_t)(k0 + bk0 + j) * NDIM + n0 + bn];
        u4 p0, p1;
        p0.x = pk2(t[0], t[1]);   p0.y = pk2(t[2], t[3]);
        p0.z = pk2(t[4], t[5]);   p0.w = pk2(t[6], t[7]);
        p1.x = pk2(t[8], t[9]);   p1.y = pk2(t[10], t[11]);
        p1.z = pk2(t[12], t[13]); p1.w = pk2(t[14], t[15]);
        *(u4*)&Bs[buf][bn][bk0]     = p0;
        *(u4*)&Bs[buf][bn][bk0 + 8] = p1;
    };

    v8f acc[2][4] = {};

    stage(0, 0);
    int cur = 0;
    for (int k0 = 0; k0 < KDIM; k0 += TK) {
        __syncthreads();                       // buf[cur] ready for all waves
        if (k0 + TK < KDIM) stage(cur ^ 1, k0 + TK);

        frag_t af[2], bfr[4];
        #pragma unroll
        for (int mt = 0; mt < 2; ++mt) {
            int row = wm + mt * 16 + r16;
            af[mt].q[0] = *(const u4*)&As[cur][row][klo];
            af[mt].q[1] = *(const u4*)&As[cur][row][16 + klo];
        }
        #pragma unroll
        for (int nt = 0; nt < 4; ++nt) {
            int col = wn + nt * 16 + r16;
            bfr[nt].q[0] = *(const u4*)&Bs[cur][col][klo];
            bfr[nt].q[1] = *(const u4*)&Bs[cur][col][16 + klo];
        }
        #pragma unroll
        for (int mt = 0; mt < 2; ++mt)
            #pragma unroll
            for (int nt = 0; nt < 4; ++nt)
                acc[mt][nt] = __builtin_amdgcn_wmma_f32_16x16x32_bf16(
                    false, af[mt].v, false, bfr[nt].v, (short)0, acc[mt][nt],
                    false, false);
        cur ^= 1;
    }

    // ---- epilogue: bias + store (base + compile-time strides) ----
    #pragma unroll
    for (int mt = 0; mt < 2; ++mt) {
        int gm0 = m0 + wm + mt * 16 + hb;       // 8 consecutive rows from here
        #pragma unroll
        for (int nt = 0; nt < 4; ++nt) {
            int gn = n0 + wn + nt * 16 + r16;
            float bs = bias[gn];
            if (MODE == 0) {
                int which = gn >> 10;           // 0=q 1=k 2=v (uniform per nt)
                int c = gn & 1023;
                int h = c >> 6, d = c & 63;
                int b = gm0 >> 11, t0 = gm0 & 2047;
                float* dst = (which == 0) ? qw : ((which == 1) ? kw : vw);
                dst += ((size_t)(b * N_HEAD + h) * SEQ + t0) * HDIM + d;
                #pragma unroll
                for (int r = 0; r < 8; ++r)     // t0+r: stride HDIM floats
                    dst[(size_t)r * HDIM] = acc[mt][nt][r] + bs;
            } else {
                float* dst = outp + (size_t)gm0 * NDIM + gn;
                #pragma unroll
                for (int r = 0; r < 8; ++r)
                    dst[(size_t)r * NDIM] = acc[mt][nt][r] + bs;
            }
        }
    }
}

// =====================================================================
// Flash attention (causal), bf16 WMMA, fp32 online softmax.
// grid.x = B*H (64), grid.y = T/64 (32). 128 threads = 4 waves.
// Each wave owns 16 query rows; K/V tiles of 64 keys staged in LDS.
// =====================================================================
#define KVP 72   // LDS pitch for K/V tiles (144B rows, 16B aligned)

__global__ __launch_bounds__(128)
void flash_attn_wmma(const float* __restrict__ qg, const float* __restrict__ kg,
                     const float* __restrict__ vg, float* __restrict__ yw)
{
    __shared__ __bf16 Kt[64][KVP];       // [key][d]
    __shared__ __bf16 Vt[HDIM][KVP];     // [d][key] (transposed for PV B-frags)
    __shared__ __bf16 Ps[4][16][64];     // per-wave P staging [row][key]

    const int bh   = blockIdx.x;          // b*H + h
    const int qblk = blockIdx.y;
    const int q0   = qblk * 64;
    const int tid  = threadIdx.x;
    const int lane = tid & 31;
    const int wave = tid >> 5;
    const int r16  = lane & 15;
    const int klo  = (lane < 16) ? 0 : 8;
    const int hb   = (lane < 16) ? 0 : 8;
    const int b    = bh >> 4;
    const int h    = bh & 15;

    // ---- load Q fragments, softmax scale (1/sqrt(64)=0.125) folded in ----
    const int qrow = q0 + wave * 16 + r16;
    const float* qp = qg + ((size_t)bh * SEQ + qrow) * HDIM;
    frag_t aq[2];
    #pragma unroll
    for (int kk = 0; kk < 2; ++kk) {
        f4 a0 = *(const f4*)(qp + kk * 32 + klo);
        f4 a1 = *(const f4*)(qp + kk * 32 + klo + 4);
        f4 a2 = *(const f4*)(qp + kk * 32 + 16 + klo);
        f4 a3 = *(const f4*)(qp + kk * 32 + 16 + klo + 4);
        aq[kk].q[0] = pk8(a0 * 0.125f, a1 * 0.125f);
        aq[kk].q[1] = pk8(a2 * 0.125f, a3 * 0.125f);
    }

    v8f o[4] = {};
    float mrow[8], lrow[8];
    #pragma unroll
    for (int r = 0; r < 8; ++r) { mrow[r] = -__builtin_inff(); lrow[r] = 0.0f; }

    const int vkey = tid & 63;            // V staging: coalesced row chunk
    const int vd0  = (tid >> 6) * 32;

    for (int kb = 0; kb <= qblk; ++kb) {
        const size_t kvbase = ((size_t)bh * SEQ + kb * 64) * HDIM;
        // ---- stage K tile: chunks of 8 floats -> packed b128 LDS stores ----
        #pragma unroll
        for (int j = 0; j < 4; ++j) {
            int c   = tid + j * 128;
            int key = c >> 3;
            int d8  = (c & 7) * 8;
            const f4* gp = (const f4*)(kg + kvbase + (size_t)key * HDIM + d8);
            *(u4*)&Kt[key][d8] = pk8(gp[0], gp[1]);
        }
        // ---- stage V transposed ----
        {
            const float* vp = vg + kvbase + (size_t)vkey * HDIM + vd0;
            #pragma unroll
            for (int j = 0; j < 32; j += 4) {
                f4 vv = *(const f4*)(vp + j);
                Vt[vd0 + j + 0][vkey] = (__bf16)vv.x;
                Vt[vd0 + j + 1][vkey] = (__bf16)vv.y;
                Vt[vd0 + j + 2][vkey] = (__bf16)vv.z;
                Vt[vd0 + j + 3][vkey] = (__bf16)vv.w;
            }
        }
        __syncthreads();

        // ---- S = Q K^T  (4 key sub-tiles of 16, d-dim 64 split in 2) ----
        v8f s[4];
        #pragma unroll
        for (int kt = 0; kt < 4; ++kt) {
            v8f a = {};
            int keyc = kt * 16 + r16;
            #pragma unroll
            for (int kk = 0; kk < 2; ++kk) {
                frag_t bk;
                bk.q[0] = *(const u4*)&Kt[keyc][kk * 32 + klo];
                bk.q[1] = *(const u4*)&Kt[keyc][kk * 32 + 16 + klo];
                a = __builtin_amdgcn_wmma_f32_16x16x32_bf16(
                    false, aq[kk].v, false, bk.v, (short)0, a, false, false);
            }
            s[kt] = a;
        }

        // ---- causal mask (diagonal block only) ----
        if (kb == qblk) {
            #pragma unroll
            for (int kt = 0; kt < 4; ++kt)
                #pragma unroll
                for (int r = 0; r < 8; ++r) {
                    int keyi = kb * 64 + kt * 16 + r16;
                    int qi   = q0 + wave * 16 + r + hb;
                    if (keyi > qi) s[kt][r] = -__builtin_inff();
                }
        }

        // ---- online softmax (rows live in 16-lane groups) ----
        #pragma unroll
        for (int r = 0; r < 8; ++r) {
            float mx = s[0][r];
            #pragma unroll
            for (int kt = 1; kt < 4; ++kt) mx = fmaxf(mx, s[kt][r]);
            #pragma unroll
            for (int off = 1; off < 16; off <<= 1)
                mx = fmaxf(mx, __shfl_xor(mx, off, 16));
            float mnew = fmaxf(mrow[r], mx);
            float corr = __expf(mrow[r] - mnew);
            float rs = 0.0f;
            #pragma unroll
            for (int kt = 0; kt < 4; ++kt) {
                float p = __expf(s[kt][r] - mnew);
                s[kt][r] = p;
                rs += p;
            }
            #pragma unroll
            for (int off = 1; off < 16; off <<= 1)
                rs += __shfl_xor(rs, off, 16);
            lrow[r] = lrow[r] * corr + rs;
            mrow[r] = mnew;
            #pragma unroll
            for (int nt = 0; nt < 4; ++nt) o[nt][r] *= corr;
        }

        // ---- stage P through LDS (wave-private region: no block barrier,
        //      intra-wave LDS ordering via DScnt suffices) ----
        #pragma unroll
        for (int kt = 0; kt < 4; ++kt)
            #pragma unroll
            for (int r = 0; r < 8; ++r)
                Ps[wave][r + hb][kt * 16 + r16] = (__bf16)s[kt][r];

        frag_t pa[2];
        #pragma unroll
        for (int kk = 0; kk < 2; ++kk) {
            pa[kk].q[0] = *(const u4*)&Ps[wave][r16][kk * 32 + klo];
            pa[kk].q[1] = *(const u4*)&Ps[wave][r16][kk * 32 + 16 + klo];
        }

        // ---- O += P V  (4 d sub-tiles, key-dim 64 split in 2) ----
        #pragma unroll
        for (int nt = 0; nt < 4; ++nt) {
            int dc = nt * 16 + r16;
            #pragma unroll
            for (int kk = 0; kk < 2; ++kk) {
                frag_t vb;
                vb.q[0] = *(const u4*)&Vt[dc][kk * 32 + klo];
                vb.q[1] = *(const u4*)&Vt[dc][kk * 32 + 16 + klo];
                o[nt] = __builtin_amdgcn_wmma_f32_16x16x32_bf16(
                    false, pa[kk].v, false, vb.v, (short)0, o[nt], false, false);
            }
        }
        __syncthreads();   // protect Kt/Vt before next iteration's staging
    }

    // ---- normalize and store y in [B,T,C] layout (C = h*64 + d) ----
    #pragma unroll
    for (int nt = 0; nt < 4; ++nt) {
        float* yp = yw + ((size_t)b * SEQ + (q0 + wave * 16 + hb)) * N_EMBD
                       + h * HDIM + nt * 16 + r16;
        #pragma unroll
        for (int r = 0; r < 8; ++r)
            yp[(size_t)r * N_EMBD] = o[nt][r] / lrow[r];
    }
}

// =====================================================================
extern "C" void kernel_launch(void* const* d_in, const int* in_sizes, int n_in,
                              void* d_out, int out_size, void* d_ws, size_t ws_size,
                              hipStream_t stream)
{
    (void)in_sizes; (void)n_in; (void)out_size; (void)ws_size;
    const float* x      = (const float*)d_in[0];
    const float* W_attn = (const float*)d_in[1];
    const float* b_attn = (const float*)d_in[2];
    const float* W_proj = (const float*)d_in[3];
    const float* b_proj = (const float*)d_in[4];
    float* out = (float*)d_out;

    const size_t NEL = (size_t)BATCH * SEQ * N_EMBD;   // 8,388,608
    float* qw = (float*)d_ws;
    float* kw = qw + NEL;
    float* vw = kw + NEL;
    float* yw = vw + NEL;

    // 1) QKV projection: [8192,1024] @ [1024,3072] + bias -> q/k/v [B,H,T,D]
    {
        dim3 grid(3 * N_EMBD / TN, MROWS / TM);   // (24, 64)
        gemm_bf16_wmma<0, 3 * N_EMBD><<<grid, 256, 0, stream>>>(
            x, W_attn, b_attn, qw, kw, vw, nullptr);
    }

    // 2) causal flash attention -> yw [B,T,C]
    {
        dim3 grid(BATCH * N_HEAD, SEQ / 64);      // (64, 32)
        flash_attn_wmma<<<grid, 128, 0, stream>>>(qw, kw, vw, yw);
    }

    // 3) output projection: [8192,1024] @ [1024,1024] + bias -> out
    {
        dim3 grid(N_EMBD / TN, MROWS / TM);       // (8, 64)
        gemm_bf16_wmma<1, N_EMBD><<<grid, 256, 0, stream>>>(
            yw, W_proj, b_proj, nullptr, nullptr, nullptr, out);
    }
}
